// Attention_63419487093402
// MI455X (gfx1250) — compile-verified, matrix operand order checked by reading, not run on hardware
//
#include <hip/hip_runtime.h>

// ---------------------------------------------------------------------------
// Transformer-XL attention block for MI455X (gfx1250, wave32, WMMA bf16).
// All large GEMMs run through v_wmma_f32_16x16x32_bf16. fp32 inputs are
// down-converted to packed bf16 (v_cvt_pk_bf16_f32) while staging into LDS;
// K/V/R projections are stored bf16 in HBM to halve traffic (23.3 TB/s roof).
// Block tile 64x128, 8 waves, 4 WMMA per wave per K-step, double-buffered LDS
// with software-pipelined global->reg->LDS staging (1 barrier per K-step).
// ---------------------------------------------------------------------------

typedef __attribute__((ext_vector_type(16))) __bf16 v16bf;
typedef __attribute__((ext_vector_type(2)))  __bf16 v2bf;
typedef __attribute__((ext_vector_type(8)))  float  v8f;

constexpr int kB = 32;          // batch
constexpr int kS = 100;         // query length
constexpr int kM = 384;         // memory length
constexpr int kE = 512;         // head dim / model dim
constexpr int kH = 10;          // heads
constexpr int kT = kS + kM;     // 484
constexpr int kD = kH * kE;     // 5120

__device__ __forceinline__ unsigned int pk_bf16(float a, float b) {
  v2bf h = { (__bf16)a, (__bf16)b };
  return __builtin_bit_cast(unsigned int, h);   // v_cvt_pk_bf16_f32
}
__device__ __forceinline__ unsigned short bf16_bits(float x) {
  union { __bf16 h; unsigned short u; } t;
  t.h = (__bf16)x; return t.u;
}

// ---------------------------------------------------------------------------
// Elementwise kernels
// ---------------------------------------------------------------------------
__global__ void concat_h_kernel(const float* __restrict__ x,
                                const float* __restrict__ mem,
                                float* __restrict__ h, long long total) {
  long long i = (long long)blockIdx.x * blockDim.x + threadIdx.x;
  if (i >= total) return;
  long long e = i % kE;
  long long t = (i / kE) % kT;
  long long b = i / ((long long)kE * kT);
  float v;
  if (t < kM) v = mem[(b * kM + t) * kE + e];
  else        v = x[(b * kS + (t - kM)) * kE + e];
  h[i] = v;
}

__global__ void pos_enc_kernel(float* __restrict__ pos, int total) {
  int i = blockIdx.x * blockDim.x + threadIdx.x;
  if (i >= total) return;
  int j = i % kE;
  int t = i / kE;
  int p = kT - 1 - t;  // enc[::-1] row flip
  float freq = powf(10000.0f, -2.0f * (float)j / (float)kE);
  float a = (float)p * freq;
  pos[i] = (j & 1) ? cosf(a) : sinf(a);
}

__global__ void add_u_kernel(float* __restrict__ q, const float* __restrict__ u,
                             long long total, long long period) {
  long long i = (long long)blockIdx.x * blockDim.x + threadIdx.x;
  if (i >= total) return;
  q[i] += u[i % period];
}

// ---------------------------------------------------------------------------
// Generic batched GEMM: C = act(A * B + bias + resid)
//   A: fp32 row-major (Mdim x Kdim), lda
//   B (template): B_BF16 dtype; B_NT: NT (Ndim rows x Kdim) vs NN (Kdim x Ndim)
//   C: fp32 or bf16, ldc
// Per-z batch offsets: off = (z/div)*shi + (z%div)*slo   (elements)
// Requires (true at all call sites): Kdim%4==0, Ndim%4==0, lda%4==0, ldb%4==0,
// batch offsets %4==0.
// ---------------------------------------------------------------------------
#define LPAD 20   // uints per LDS row (16 data + 4 pad; %4==0 -> b128 aligned)

template <int B_BF16, int B_NT>
__global__ void __launch_bounds__(256)
gemm_wmma_kernel(const float* __restrict__ Ag,
                 long long aDiv, long long aShi, long long aSlo, int lda,
                 const void* __restrict__ Bg,
                 long long bDiv, long long bShi, long long bSlo, int ldb,
                 const float* __restrict__ bias,
                 const float* __restrict__ resid, int ldres,
                 void* __restrict__ Cg, int c_bf16,
                 long long cDiv, long long cShi, long long cSlo, int ldc,
                 int Mdim, int Ndim, int Kdim, int relu) {
  __shared__ unsigned int sA32[2][64 * LPAD];    // [buf][m][k/2]
  __shared__ unsigned int sB32[2][128 * LPAD];   // [buf][n][k/2]

  const int tid  = threadIdx.x;
  const int lane = tid & 31;
  const int wv   = tid >> 5;        // 0..7
  const int wm   = wv >> 1;         // 0..3  : M subtile
  const int wn   = wv & 1;          // 0..1  : N half
  const long long z = blockIdx.z;

  const float* A = Ag + (size_t)(z / aDiv) * aShi + (size_t)(z % aDiv) * aSlo;
  const size_t bOff = (size_t)(z / bDiv) * bShi + (size_t)(z % bDiv) * bSlo;
  const size_t cOff = (size_t)(z / cDiv) * cShi + (size_t)(z % cDiv) * cSlo;

  const int mTile = blockIdx.y * 64;
  const int nTile = blockIdx.x * 128;

  const int half = lane >> 4;       // 0/1: lane group
  const int l15  = lane & 15;
  const int mrow = wm * 16 + l15;   // A fragment row within block tile

  // ---- register staging state ----
  float4 ra[2];  bool okA[2];
  float4 rbf[4]; uint2 rbu[4]; bool okB[4];

  // Phase 1: issue global loads for K-tile `kb` (unconditional, clamped).
  auto loadPhase = [&](int kb) {
#pragma unroll
    for (int i = 0; i < 2; ++i) {           // A: 64x32, 512 float4, 2/thread
      int c = tid + i * 256;
      int m = c >> 3, k4 = (c & 7) << 2;
      int gm = mTile + m, gk = kb + k4;
      okA[i] = (gm < Mdim) && (gk < Kdim);
      int gmc = (gm < Mdim) ? gm : (Mdim - 1);
      int gkc = (gk < Kdim) ? gk : 0;
      ra[i] = *(const float4*)(A + (size_t)gmc * lda + gkc);
    }
    if constexpr (B_NT) {                   // B rows indexed by n: [n][k]
#pragma unroll
      for (int i = 0; i < 4; ++i) {
        int c = tid + i * 256;
        int n = c >> 3, k4 = (c & 7) << 2;
        int gn = nTile + n, gk = kb + k4;
        okB[i] = (gn < Ndim) && (gk < Kdim);
        int gnc = (gn < Ndim) ? gn : (Ndim - 1);
        int gkc = (gk < Kdim) ? gk : 0;
        size_t off = bOff + (size_t)gnc * ldb + gkc;   // element offset, %4==0
        if constexpr (B_BF16) rbu[i] = *(const uint2*)((const unsigned int*)Bg + (off >> 1));
        else                  rbf[i] = *(const float4*)((const float*)Bg + off);
      }
    } else {                                // B is K x N
#pragma unroll
      for (int i = 0; i < 4; ++i) {
        int c = tid + i * 256;
        int k = c >> 5, n4 = (c & 31) << 2;
        int gk = kb + k, gn = nTile + n4;
        okB[i] = (gk < Kdim) && (gn < Ndim);
        int gkc = (gk < Kdim) ? gk : 0;
        int gnc = (gn < Ndim) ? gn : (Ndim - 4);
        size_t off = bOff + (size_t)gkc * ldb + gnc;
        if constexpr (B_BF16) rbu[i] = *(const uint2*)((const unsigned short*)Bg + off);
        else                  rbf[i] = *(const float4*)((const float*)Bg + off);
      }
    }
  };

  // Phase 2: convert + store staged registers into LDS buffer `buf`.
  auto storePhase = [&](int buf) {
#pragma unroll
    for (int i = 0; i < 2; ++i) {
      int c = tid + i * 256;
      int m = c >> 3, k4 = (c & 7) << 2;
      float4 v = ra[i];
      if (!okA[i]) { v.x = 0.f; v.y = 0.f; v.z = 0.f; v.w = 0.f; }
      sA32[buf][m * LPAD + (k4 >> 1)]     = pk_bf16(v.x, v.y);
      sA32[buf][m * LPAD + (k4 >> 1) + 1] = pk_bf16(v.z, v.w);
    }
    if constexpr (B_NT) {
#pragma unroll
      for (int i = 0; i < 4; ++i) {
        int c = tid + i * 256;
        int n = c >> 3, k4 = (c & 7) << 2;
        unsigned int u0, u1;
        if constexpr (B_BF16) { u0 = rbu[i].x; u1 = rbu[i].y; }
        else { u0 = pk_bf16(rbf[i].x, rbf[i].y); u1 = pk_bf16(rbf[i].z, rbf[i].w); }
        if (!okB[i]) { u0 = 0u; u1 = 0u; }
        sB32[buf][n * LPAD + (k4 >> 1)]     = u0;
        sB32[buf][n * LPAD + (k4 >> 1) + 1] = u1;
      }
    } else {
      unsigned short* sB16 = (unsigned short*)&sB32[buf][0];
#pragma unroll
      for (int i = 0; i < 4; ++i) {
        int c = tid + i * 256;
        int k = c >> 5, n4 = (c & 31) << 2;
        unsigned short h0, h1, h2, h3;
        if constexpr (B_BF16) {
          h0 = (unsigned short)(rbu[i].x & 0xffffu);
          h1 = (unsigned short)(rbu[i].x >> 16);
          h2 = (unsigned short)(rbu[i].y & 0xffffu);
          h3 = (unsigned short)(rbu[i].y >> 16);
        } else {
          h0 = bf16_bits(rbf[i].x); h1 = bf16_bits(rbf[i].y);
          h2 = bf16_bits(rbf[i].z); h3 = bf16_bits(rbf[i].w);
        }
        if (!okB[i]) { h0 = 0; h1 = 0; h2 = 0; h3 = 0; }
        sB16[(n4 + 0) * (2 * LPAD) + k] = h0;
        sB16[(n4 + 1) * (2 * LPAD) + k] = h1;
        sB16[(n4 + 2) * (2 * LPAD) + k] = h2;
        sB16[(n4 + 3) * (2 * LPAD) + k] = h3;
      }
    }
  };

  v8f acc[4];
#pragma unroll
  for (int nn = 0; nn < 4; ++nn) acc[nn] = (v8f){0.f,0.f,0.f,0.f,0.f,0.f,0.f,0.f};

  // ---- prologue ----
  loadPhase(0);
  storePhase(0);
  __syncthreads();

  const int iters = (Kdim + 31) >> 5;
  for (int it = 0; it < iters; ++it) {
    const int cur = it & 1;
    const bool more = (it + 1) < iters;
    if (more) loadPhase((it + 1) * 32);     // global loads in flight...

    // ---- fragments + WMMA on current LDS buffer (hides global latency) ----
    // v16bf element 2j/2j+1 <-> uint j; uint j<4 -> pair half*4+j,
    // uint j>=4 -> pair 8+half*4+(j-4)   (ISA 7.12.2 16-bit layout)
    union FragU { unsigned int u[8]; v16bf v; };
    FragU fa;
    {
      const unsigned int* ar = &sA32[cur][mrow * LPAD + half * 4];
      fa.u[0] = ar[0]; fa.u[1] = ar[1]; fa.u[2] = ar[2]; fa.u[3] = ar[3];
      fa.u[4] = ar[8]; fa.u[5] = ar[9]; fa.u[6] = ar[10]; fa.u[7] = ar[11];
    }
#pragma unroll
    for (int nn = 0; nn < 4; ++nn) {
      FragU fb;
      const unsigned int* br = &sB32[cur][(wn * 64 + nn * 16 + l15) * LPAD + half * 4];
      fb.u[0] = br[0]; fb.u[1] = br[1]; fb.u[2] = br[2]; fb.u[3] = br[3];
      fb.u[4] = br[8]; fb.u[5] = br[9]; fb.u[6] = br[10]; fb.u[7] = br[11];
      acc[nn] = __builtin_amdgcn_wmma_f32_16x16x32_bf16(
          false, fa.v, false, fb.v, (short)0, acc[nn], false, false);
    }

    if (more) storePhase(cur ^ 1);          // ...landed by now; fill next buffer
    __syncthreads();
  }

  // ---- epilogue: C row = half*8 + j, col = lane&15 (ISA 7.12.2) ----
#pragma unroll
  for (int nn = 0; nn < 4; ++nn) {
    int gn = nTile + wn * 64 + nn * 16 + l15;
#pragma unroll
    for (int j = 0; j < 8; ++j) {
      int gm = mTile + wm * 16 + half * 8 + j;
      if (gm < Mdim && gn < Ndim) {
        float v = acc[nn][j];
        if (bias)  v += bias[gn];
        if (resid) v += resid[(size_t)gm * ldres + gn];
        if (relu)  v = fmaxf(v, 0.f);
        size_t co = cOff + (size_t)gm * ldc + gn;
        if (c_bf16) ((__bf16*)Cg)[co] = (__bf16)v;
        else        ((float*)Cg)[co]  = v;
      }
    }
  }
}

// ---------------------------------------------------------------------------
// Fused: att[s,t] = softmax_t( mask * (qk_uk + qr[shifted] + vr) / sqrt(E) )
// One block (128 threads) per (b,h,s) row; in-place on sc.
// Masked entries are 0 (not -inf) before softmax — faithful to reference.
// ---------------------------------------------------------------------------
__global__ void __launch_bounds__(128)
score_softmax_kernel(float* __restrict__ sc, const float* __restrict__ qr,
                     const float* __restrict__ vr) {
  const int row = blockIdx.x;       // z*S + s, z = b*H + h
  const int z   = row / kS;
  const int s   = row - z * kS;
  const int h   = z % kH;
  const int tid = threadIdx.x;
  const size_t base  = (size_t)z * kS * kT + (size_t)s * kT;
  const size_t vbase = ((size_t)h * kS + s) * kT;

  __shared__ float buf[kT];
  __shared__ float red[128];
  const float scale = 0.04419417382415922f;  // 1/sqrt(512)

  float lmax = -3.0e38f;
  for (int t = tid; t < kT; t += 128) {
    int sh = (kS - 1 - s + t) % kT;  // circulant shift for bterm
    float v = sc[base + t] + qr[base + sh] + vr[vbase + t];
    v = (t <= s + kM) ? v * scale : 0.0f;
    buf[t] = v;
    lmax = fmaxf(lmax, v);
  }
  red[tid] = lmax; __syncthreads();
  for (int o = 64; o > 0; o >>= 1) {
    if (tid < o) red[tid] = fmaxf(red[tid], red[tid + o]);
    __syncthreads();
  }
  const float mx = red[0]; __syncthreads();

  float ls = 0.f;
  for (int t = tid; t < kT; t += 128) {
    float e = expf(buf[t] - mx);
    buf[t] = e;
    ls += e;
  }
  red[tid] = ls; __syncthreads();
  for (int o = 64; o > 0; o >>= 1) {
    if (tid < o) red[tid] += red[tid + o];
    __syncthreads();
  }
  const float inv = 1.0f / red[0]; __syncthreads();

  for (int t = tid; t < kT; t += 128) sc[base + t] = buf[t] * inv;
}

// ---------------------------------------------------------------------------
// LayerNorm over E=512 per row (population variance, eps 1e-5)
// ---------------------------------------------------------------------------
__global__ void __launch_bounds__(128)
layernorm_kernel(const float* __restrict__ in, const float* __restrict__ g,
                 const float* __restrict__ b, float* __restrict__ out) {
  const int row = blockIdx.x;
  const int tid = threadIdx.x;
  __shared__ float buf[kE];
  __shared__ float red[128];

  float s1 = 0.f, s2 = 0.f;
  for (int e = tid; e < kE; e += 128) {
    float v = in[(size_t)row * kE + e];
    buf[e] = v;
    s1 += v; s2 += v * v;
  }
  red[tid] = s1; __syncthreads();
  for (int o = 64; o > 0; o >>= 1) { if (tid < o) red[tid] += red[tid + o]; __syncthreads(); }
  const float mean = red[0] * (1.0f / kE); __syncthreads();
  red[tid] = s2; __syncthreads();
  for (int o = 64; o > 0; o >>= 1) { if (tid < o) red[tid] += red[tid + o]; __syncthreads(); }
  const float var = red[0] * (1.0f / kE) - mean * mean;
  const float rstd = rsqrtf(var + 1e-5f);
  __syncthreads();
  for (int e = tid; e < kE; e += 128)
    out[(size_t)row * kE + e] = (buf[e] - mean) * rstd * g[e] + b[e];
}

// ---------------------------------------------------------------------------
// Host launcher
// ---------------------------------------------------------------------------
static inline void gemm(hipStream_t st, int Mdim, int Ndim, int Kdim, int batches,
                        const float* A, long long aDiv, long long aShi, long long aSlo, int lda,
                        const void* B, int b_bf16, int b_nt,
                        long long bDiv, long long bShi, long long bSlo, int ldb,
                        const float* bias, const float* resid, int ldres,
                        void* C, int c_bf16,
                        long long cDiv, long long cShi, long long cSlo, int ldc, int relu) {
  dim3 g((Ndim + 127) / 128, (Mdim + 63) / 64, batches);
  dim3 blk(256);
#define GEMM_ARGS A, aDiv, aShi, aSlo, lda, B, bDiv, bShi, bSlo, ldb, \
                  bias, resid, ldres, C, c_bf16, cDiv, cShi, cSlo, ldc, \
                  Mdim, Ndim, Kdim, relu
  if (b_nt) {
    if (b_bf16) gemm_wmma_kernel<1, 1><<<g, blk, 0, st>>>(GEMM_ARGS);
    else        gemm_wmma_kernel<0, 1><<<g, blk, 0, st>>>(GEMM_ARGS);
  } else {
    if (b_bf16) gemm_wmma_kernel<1, 0><<<g, blk, 0, st>>>(GEMM_ARGS);
    else        gemm_wmma_kernel<0, 0><<<g, blk, 0, st>>>(GEMM_ARGS);
  }
#undef GEMM_ARGS
}

extern "C" void kernel_launch(void* const* d_in, const int* in_sizes, int n_in,
                              void* d_out, int out_size, void* d_ws, size_t ws_size,
                              hipStream_t stream) {
  (void)in_sizes; (void)n_in; (void)out_size; (void)ws_size;
  const float* x    = (const float*)d_in[0];
  const float* mem  = (const float*)d_in[1];
  const float* u    = (const float*)d_in[2];
  const float* vpar = (const float*)d_in[3];
  const float* Wq   = (const float*)d_in[4];
  const float* bq   = (const float*)d_in[5];
  const float* Wke  = (const float*)d_in[6];
  const float* bke  = (const float*)d_in[7];
  const float* Wkr  = (const float*)d_in[8];
  const float* bkr  = (const float*)d_in[9];
  const float* Wv   = (const float*)d_in[10];
  const float* bv   = (const float*)d_in[11];
  const float* Wmlp = (const float*)d_in[12];
  const float* bmlp = (const float*)d_in[13];
  const float* ln_g = (const float*)d_in[14];
  const float* ln_b = (const float*)d_in[15];
  const float* W1   = (const float*)d_in[16];
  const float* b1   = (const float*)d_in[17];
  const float* W2   = (const float*)d_in[18];
  const float* b2   = (const float*)d_in[19];
  float* out = (float*)d_out;

  char* ws = (char*)d_ws;
  auto AL = [](size_t v) { return (v + 255) & ~(size_t)255; };
  size_t cur = 0;
  auto take = [&](size_t bytes) { void* p = ws + cur; cur = AL(cur + bytes); return p; };

  float*  hbuf  = (float*)take((size_t)kB * kT * kE * 4);      // concat(mem,x)
  float*  posb  = (float*)take((size_t)kT * kE * 4);           // sinusoid pos
  float*  qbuf  = (float*)take((size_t)kB * kS * kD * 4);      // q proj (later q+u)
  __bf16* kbuf  = (__bf16*)take((size_t)kB * kT * kD * 2);     // k proj bf16
  __bf16* vbuf  = (__bf16*)take((size_t)kB * kT * kD * 2);     // v proj bf16
  __bf16* rbuf  = (__bf16*)take((size_t)kT * kD * 2);          // r proj bf16
  float*  vrbuf = (float*)take((size_t)kH * kS * kT * 4);      // v @ r^T
  float*  qrbuf = (float*)take((size_t)kB * kH * kS * kT * 4); // q @ r^T
  float*  scbuf = (float*)take((size_t)kB * kH * kS * kT * 4); // scores / att
  // reuse (dependency-safe):
  float* ctxbuf = (float*)kbuf;                                // k dead after scores
  float* out0   = hbuf;                                        // h dead after k/v proj
  float* lnbuf  = hbuf + (size_t)kB * kS * kE;
  float* ffn1   = qrbuf;                                       // qr dead after softmax

  const long long SE = (long long)kS * kE;
  const long long TE = (long long)kT * kE;
  const long long ST = (long long)kS * kT;

  // 1) h = concat(mem, x)
  {
    long long total = (long long)kB * kT * kE;
    concat_h_kernel<<<dim3((unsigned)((total + 255) / 256)), dim3(256), 0, stream>>>(
        x, mem, hbuf, total);
  }
  // 2) positional encoding
  {
    int total = kT * kE;
    pos_enc_kernel<<<dim3((total + 255) / 256), dim3(256), 0, stream>>>(posb, total);
  }
  // 3) projections
  gemm(stream, kB * kS, kD, kE, 1, x,    1, 0, 0, kE, Wq,  0, 0, 1, 0, 0, kD,
       bq,  nullptr, 0, qbuf, 0, 1, 0, 0, kD, 0);
  gemm(stream, kB * kT, kD, kE, 1, hbuf, 1, 0, 0, kE, Wke, 0, 0, 1, 0, 0, kD,
       bke, nullptr, 0, kbuf, 1, 1, 0, 0, kD, 0);
  gemm(stream, kB * kT, kD, kE, 1, hbuf, 1, 0, 0, kE, Wv,  0, 0, 1, 0, 0, kD,
       bv,  nullptr, 0, vbuf, 1, 1, 0, 0, kD, 0);
  gemm(stream, kT,     kD, kE, 1, posb, 1, 0, 0, kE, Wkr, 0, 0, 1, 0, 0, kD,
       bkr, nullptr, 0, rbuf, 1, 1, 0, 0, kD, 0);
  // 4) vr[h] = v_param[h] @ r[h]^T      (batch z = h)
  gemm(stream, kS, kT, kE, kH, vpar, 1, SE, 0, kE, rbuf, 1, 1, 1, TE, 0, kE,
       nullptr, nullptr, 0, vrbuf, 0, 1, ST, 0, kT, 0);
  // 5) qr[b,h] = q[b,h] @ r[h]^T        (batch z = b*H+h)
  gemm(stream, kS, kT, kE, kB * kH, qbuf, 1, SE, 0, kE, rbuf, 1, 1, kH, 0, TE, kE,
       nullptr, nullptr, 0, qrbuf, 0, 1, ST, 0, kT, 0);
  // 6) q += u (broadcast) — after qr
  {
    long long total = (long long)kB * kS * kD;
    long long period = (long long)kH * kS * kE;
    add_u_kernel<<<dim3((unsigned)((total + 255) / 256)), dim3(256), 0, stream>>>(
        qbuf, u, total, period);
  }
  // 7) scores[b,h] = (q+u)[b,h] @ k[b,h]^T
  gemm(stream, kS, kT, kE, kB * kH, qbuf, 1, SE, 0, kE, kbuf, 1, 1, 1, TE, 0, kE,
       nullptr, nullptr, 0, scbuf, 0, 1, ST, 0, kT, 0);
  // 8) shift + mask + scale + softmax (in-place)
  score_softmax_kernel<<<dim3(kB * kH * kS), dim3(128), 0, stream>>>(scbuf, qrbuf, vrbuf);
  // 9) ctx[b,h] = att[b,h] @ val[b,h]   -> layout (B, S, H*E)
  gemm(stream, kS, kE, kT, kB * kH, scbuf, 1, ST, 0, kT, vbuf, 1, 0, 1, TE, 0, kE,
       nullptr, nullptr, 0, ctxbuf, 0, kH, (long long)kS * kD, kE, kD, 0);
  // 10) out0 = ctx @ Wmlp + bmlp + x
  gemm(stream, kB * kS, kE, kD, 1, ctxbuf, 1, 0, 0, kD, Wmlp, 0, 0, 1, 0, 0, kE,
       bmlp, x, kE, out0, 0, 1, 0, 0, kE, 0);
  // 11) LayerNorm
  layernorm_kernel<<<dim3(kB * kS), dim3(128), 0, stream>>>(out0, ln_g, ln_b, lnbuf);
  // 12) ffn1 = relu(ln @ W1 + b1)
  gemm(stream, kB * kS, 4 * kE, kE, 1, lnbuf, 1, 0, 0, kE, W1, 0, 0, 1, 0, 0, 4 * kE,
       b1, nullptr, 0, ffn1, 0, 1, 0, 0, 4 * kE, 1);
  // 13) out = ffn1 @ W2 + b2
  gemm(stream, kB * kS, kE, 4 * kE, 1, ffn1, 1, 0, 0, 4 * kE, W2, 0, 0, 1, 0, 0, kE,
       b2, nullptr, 0, out, 0, 1, 0, 0, kE, 0);
}
